// TIGER_68985764708363
// MI455X (gfx1250) — compile-verified, matrix operand order checked by reading, not run on hardware
//
#include <hip/hip_runtime.h>
#include <hip/hip_bf16.h>

// ---- problem constants (from reference) ----
#define N_NODES 200000
#define MEMD    128
#define EVT     128
#define TDIM    100
#define NEV     100000
#define MSG     484       // 2*MEM + EVT + TDIM
#define KPAD    512       // MSG padded to multiple of 32 for WMMA K-loop
#define GATES   384       // 3*MEM
#define ROWS    32        // nodes per block (two 16-row WMMA M-tiles)

typedef __attribute__((ext_vector_type(16))) __bf16 bf16x16;
typedef __attribute__((ext_vector_type(8)))  float  f32x8;

__device__ __forceinline__ float sigmoidf_(float x) { return 1.0f / (1.0f + __expf(-x)); }

// ------------------------------------------------------------------
// K0: zero the per-node packed keys
// ------------------------------------------------------------------
__global__ void tgn_init_keys(unsigned long long* __restrict__ keys) {
    int i = blockIdx.x * blockDim.x + threadIdx.x;
    if (i < N_NODES) keys[i] = 0ull;
}

// ------------------------------------------------------------------
// K1: 'last' aggregator as packed 64-bit atomicMax.
// key = (float_bits(t) << 32) | (eventIdx+1).  t >= 0 so float bits are
// monotone; low word reproduces the reference tie-break (max event index)
// and doubles as has_msg (nonzero).
// ------------------------------------------------------------------
__global__ void tgn_scatter_max(const int* __restrict__ src,
                                const int* __restrict__ dst,
                                const float* __restrict__ t,
                                unsigned long long* __restrict__ keys) {
    int i = blockIdx.x * blockDim.x + threadIdx.x;
    if (i >= 2 * NEV) return;
    int e = (i < NEV) ? i : (i - NEV);
    int v = (i < NEV) ? src[e] : dst[e];
    unsigned int tb = __float_as_uint(t[e]);
    unsigned long long key =
        ((unsigned long long)tb << 32) | (unsigned long long)(unsigned int)(i + 1);
    atomicMax(&keys[v], key);
}

// ------------------------------------------------------------------
// K2: weight prep -> bf16, column-major (N-major, contiguous K), zero-padded.
// wB_ih: [384][512], wB_hh: [384][128]. Stays resident in the 192MB L2.
// ------------------------------------------------------------------
__global__ void tgn_prep_weights(const float* __restrict__ w_ih,
                                 const float* __restrict__ w_hh,
                                 __bf16* __restrict__ wB_ih,
                                 __bf16* __restrict__ wB_hh) {
    int i = blockIdx.x * blockDim.x + threadIdx.x;
    if (i < GATES * KPAD) {
        int n = i / KPAD, k = i % KPAD;
        float v = (k < MSG) ? w_ih[k * GATES + n] : 0.0f;
        wB_ih[n * KPAD + k] = (__bf16)v;
    } else {
        int j = i - GATES * KPAD;
        if (j < GATES * MEMD) {
            int n = j / MEMD, k = j % MEMD;
            wB_hh[n * MEMD + k] = (__bf16)w_hh[k * GATES + n];
        }
    }
}

// ------------------------------------------------------------------
// K3: fused message-build + gather + dual GEMM (WMMA bf16) + GRU epilogue.
// One block = 32 nodes (two 16-row M-tiles), 8 waves. Wave w owns 16-col
// tiles {w, w+8, w+16} so each lane holds r/z/n gate values for the same
// (node, unit). Each B fragment (L2-resident weights) feeds TWO WMMAs.
// K-loops deliberately NOT unrolled: keeps live VGPRs < 256 (no spills);
// latency is hidden by the 8 resident wave32 waves, not by unrolling.
// ------------------------------------------------------------------
__global__ __launch_bounds__(256)
void tgn_gru_update(const float* __restrict__ memory,
                    const float* __restrict__ last_update,
                    const float* __restrict__ t,
                    const float* __restrict__ event_feat,
                    const float* __restrict__ s_embeds,
                    const float* __restrict__ d_embeds,
                    const float* __restrict__ time_w,
                    const float* __restrict__ time_b,
                    const float* __restrict__ b_ih,
                    const float* __restrict__ b_hh,
                    const __bf16* __restrict__ wB_ih,
                    const __bf16* __restrict__ wB_hh,
                    const unsigned long long* __restrict__ keys,
                    float* __restrict__ mem_out) {
    __shared__ __align__(32) __bf16 xA[ROWS][KPAD];   // 32KB message tile (bf16)
    __shared__ __align__(32) __bf16 hA[ROWS][MEMD];   // 8KB  memory tile (bf16)
    __shared__ __align__(32) float  hF[ROWS][MEMD];   // 16KB memory tile (f32)
    __shared__ int hasFlag[ROWS];

    const int tid  = threadIdx.x;
    const int base = blockIdx.x * ROWS;
    const int r    = tid >> 3;     // row 0..31 (8 threads per row)
    const int sub  = tid & 7;

    const int v = base + r;
    const unsigned long long key = keys[v];
    const int has = ((unsigned int)key) != 0u;

    // ---- stage memory rows (f32 + bf16) ----
    {
        const float* mrow = memory + (long long)v * MEMD;
        const int c0 = sub * 16;
        #pragma unroll
        for (int c = 0; c < 16; ++c) {
            float h = mrow[c0 + c];
            hF[r][c0 + c] = h;
            hA[r][c0 + c] = (__bf16)h;
        }
    }

    // ---- build the 484-wide message row on the fly (gather + time enc) ----
    {
        const int c0 = sub * 64;
        if (has) {
            unsigned int p = (unsigned int)key - 1u;
            int flip = (p >= NEV);
            int e = flip ? (int)(p - NEV) : (int)p;
            const float* selfe  = flip ? d_embeds : s_embeds;
            const float* othere = flip ? s_embeds : d_embeds;
            float dt = t[e] - last_update[v];
            #pragma unroll 4
            for (int c = c0; c < c0 + 64; ++c) {
                float val;
                if (c < 128)      val = selfe [(long long)e * MEMD + c];
                else if (c < 256) val = othere[(long long)e * MEMD + (c - 128)];
                else if (c < 384) val = event_feat[(long long)e * EVT + (c - 256)];
                else if (c < MSG) val = __cosf(dt * time_w[c - 384] + time_b[c - 384]);
                else              val = 0.0f;
                xA[r][c] = (__bf16)val;
            }
        } else {
            #pragma unroll
            for (int c = c0; c < c0 + 64; ++c) xA[r][c] = (__bf16)0.0f;
        }
        if (sub == 0) hasFlag[r] = has;
    }

    __syncthreads();

    // all-empty tile: plain copy-through, skip the GEMMs
    int any = 0;
    #pragma unroll
    for (int i = 0; i < ROWS; ++i) any |= hasFlag[i];
    if (!any) {
        const int c0 = sub * 16;
        float* orow = mem_out + (long long)v * MEMD;
        #pragma unroll
        for (int c = 0; c < 16; ++c) orow[c0 + c] = hF[r][c0 + c];
        return;
    }

    const int lane  = tid & 31;            // wave32
    const int wave  = tid >> 5;            // 8 waves
    const int nloc  = lane & 15;           // column within 16-wide tile
    const int khalf = (lane >> 4) << 4;    // 0 or 16 (K split across lane halves)
    const int col   = (wave << 4) + nloc;  // 0..127 : unit within each gate

    __builtin_prefetch(b_ih + col, 0, 3);
    __builtin_prefetch(b_hh + col, 0, 3);

    // accumulators: [gate][Mtile]
    f32x8 aI00 = {}, aI10 = {}, aI20 = {};   // gi gates r/z/n, rows 0..15
    f32x8 aI01 = {}, aI11 = {}, aI21 = {};   // gi gates r/z/n, rows 16..31
    f32x8 aH00 = {}, aH10 = {}, aH20 = {};   // gh gates r/z/n, rows 0..15
    f32x8 aH01 = {}, aH11 = {}, aH21 = {};   // gh gates r/z/n, rows 16..31

    // gi = x @ W_ih  (K = 512 padded); each B fragment reused for both M-tiles
    {
        const __bf16* w0 = wB_ih + (long long)(0 * 128 + col) * KPAD;
        const __bf16* w1 = wB_ih + (long long)(1 * 128 + col) * KPAD;
        const __bf16* w2 = wB_ih + (long long)(2 * 128 + col) * KPAD;
        #pragma unroll 1
        for (int kt = 0; kt < KPAD; kt += 32) {
            bf16x16 aLo = *(const bf16x16*)&xA[nloc][kt + khalf];
            bf16x16 aHi = *(const bf16x16*)&xA[16 + nloc][kt + khalf];
            bf16x16 b0  = *(const bf16x16*)&w0[kt + khalf];
            bf16x16 b1  = *(const bf16x16*)&w1[kt + khalf];
            bf16x16 b2  = *(const bf16x16*)&w2[kt + khalf];
            aI00 = __builtin_amdgcn_wmma_f32_16x16x32_bf16(false, aLo, false, b0, (short)0, aI00, false, false);
            aI01 = __builtin_amdgcn_wmma_f32_16x16x32_bf16(false, aHi, false, b0, (short)0, aI01, false, false);
            aI10 = __builtin_amdgcn_wmma_f32_16x16x32_bf16(false, aLo, false, b1, (short)0, aI10, false, false);
            aI11 = __builtin_amdgcn_wmma_f32_16x16x32_bf16(false, aHi, false, b1, (short)0, aI11, false, false);
            aI20 = __builtin_amdgcn_wmma_f32_16x16x32_bf16(false, aLo, false, b2, (short)0, aI20, false, false);
            aI21 = __builtin_amdgcn_wmma_f32_16x16x32_bf16(false, aHi, false, b2, (short)0, aI21, false, false);
        }
    }
    // gh = h @ W_hh  (K = 128)
    {
        const __bf16* w0 = wB_hh + (long long)(0 * 128 + col) * MEMD;
        const __bf16* w1 = wB_hh + (long long)(1 * 128 + col) * MEMD;
        const __bf16* w2 = wB_hh + (long long)(2 * 128 + col) * MEMD;
        #pragma unroll 1
        for (int kt = 0; kt < MEMD; kt += 32) {
            bf16x16 aLo = *(const bf16x16*)&hA[nloc][kt + khalf];
            bf16x16 aHi = *(const bf16x16*)&hA[16 + nloc][kt + khalf];
            bf16x16 b0  = *(const bf16x16*)&w0[kt + khalf];
            bf16x16 b1  = *(const bf16x16*)&w1[kt + khalf];
            bf16x16 b2  = *(const bf16x16*)&w2[kt + khalf];
            aH00 = __builtin_amdgcn_wmma_f32_16x16x32_bf16(false, aLo, false, b0, (short)0, aH00, false, false);
            aH01 = __builtin_amdgcn_wmma_f32_16x16x32_bf16(false, aHi, false, b0, (short)0, aH01, false, false);
            aH10 = __builtin_amdgcn_wmma_f32_16x16x32_bf16(false, aLo, false, b1, (short)0, aH10, false, false);
            aH11 = __builtin_amdgcn_wmma_f32_16x16x32_bf16(false, aHi, false, b1, (short)0, aH11, false, false);
            aH20 = __builtin_amdgcn_wmma_f32_16x16x32_bf16(false, aLo, false, b2, (short)0, aH20, false, false);
            aH21 = __builtin_amdgcn_wmma_f32_16x16x32_bf16(false, aHi, false, b2, (short)0, aH21, false, false);
        }
    }

    // ---- GRU gate combine (register-local per lane) + masked store ----
    const float bir = b_ih[col],       bhr = b_hh[col];
    const float biz = b_ih[128 + col], bhz = b_hh[128 + col];
    const float bin = b_ih[256 + col], bhn = b_hh[256 + col];
    const int mbase = (lane >> 4) << 3;   // C layout: vgpr j -> row j (+8 for hi lanes)
    #pragma unroll
    for (int j = 0; j < 8; ++j) {        // M-tile 0: rows 0..15
        int m = mbase + j;
        float h  = hF[m][col];
        float rg = sigmoidf_(aI00[j] + bir + aH00[j] + bhr);
        float zg = sigmoidf_(aI10[j] + biz + aH10[j] + bhz);
        float ng = tanhf(aI20[j] + bin + rg * (aH20[j] + bhn));
        float o  = (1.0f - zg) * ng + zg * h;
        mem_out[(long long)(base + m) * MEMD + col] = hasFlag[m] ? o : h;
    }
    #pragma unroll
    for (int j = 0; j < 8; ++j) {        // M-tile 1: rows 16..31
        int m = 16 + mbase + j;
        float h  = hF[m][col];
        float rg = sigmoidf_(aI01[j] + bir + aH01[j] + bhr);
        float zg = sigmoidf_(aI11[j] + biz + aH11[j] + bhz);
        float ng = tanhf(aI21[j] + bin + rg * (aH21[j] + bhn));
        float o  = (1.0f - zg) * ng + zg * h;
        mem_out[(long long)(base + m) * MEMD + col] = hasFlag[m] ? o : h;
    }
}

// ------------------------------------------------------------------
// K4: last_update output
// ------------------------------------------------------------------
__global__ void tgn_t_out(const unsigned long long* __restrict__ keys,
                          const float* __restrict__ last_update,
                          float* __restrict__ t_out) {
    int v = blockIdx.x * blockDim.x + threadIdx.x;
    if (v >= N_NODES) return;
    unsigned long long key = keys[v];
    int has = ((unsigned int)key) != 0u;
    float tm = __uint_as_float((unsigned int)(key >> 32));
    t_out[v] = has ? tm : last_update[v];
}

extern "C" void kernel_launch(void* const* d_in, const int* in_sizes, int n_in,
                              void* d_out, int out_size, void* d_ws, size_t ws_size,
                              hipStream_t stream) {
    (void)in_sizes; (void)n_in; (void)out_size; (void)ws_size;
    const float* memory      = (const float*)d_in[0];
    const float* last_update = (const float*)d_in[1];
    const float* t           = (const float*)d_in[2];
    const float* event_feat  = (const float*)d_in[3];
    const float* s_embeds    = (const float*)d_in[4];
    const float* d_embeds    = (const float*)d_in[5];
    const float* time_w      = (const float*)d_in[6];
    const float* time_b      = (const float*)d_in[7];
    const float* w_ih        = (const float*)d_in[8];
    const float* w_hh        = (const float*)d_in[9];
    const float* b_ih        = (const float*)d_in[10];
    const float* b_hh        = (const float*)d_in[11];
    const int*   src         = (const int*)d_in[12];
    const int*   dst         = (const int*)d_in[13];

    float* mem_out = (float*)d_out;                              // [N,128]
    float* t_out   = (float*)d_out + (long long)N_NODES * MEMD;  // [N]

    // workspace carve-up (all 32B-aligned)
    char* ws = (char*)d_ws;
    unsigned long long* keys = (unsigned long long*)ws;          // N*8 = 1.6MB
    __bf16* wB_ih = (__bf16*)(ws + (size_t)N_NODES * 8);         // 384*512*2
    __bf16* wB_hh = (__bf16*)(ws + (size_t)N_NODES * 8 + (size_t)GATES * KPAD * 2);

    tgn_init_keys<<<(N_NODES + 255) / 256, 256, 0, stream>>>(keys);
    tgn_scatter_max<<<(2 * NEV + 255) / 256, 256, 0, stream>>>(src, dst, t, keys);
    tgn_prep_weights<<<(GATES * (KPAD + MEMD) + 255) / 256, 256, 0, stream>>>(w_ih, w_hh, wB_ih, wB_hh);
    tgn_gru_update<<<N_NODES / ROWS, 256, 0, stream>>>(
        memory, last_update, t, event_feat, s_embeds, d_embeds,
        time_w, time_b, b_ih, b_hh, wB_ih, wB_hh, keys, mem_out);
    tgn_t_out<<<(N_NODES + 255) / 256, 256, 0, stream>>>(keys, last_update, t_out);
}
